// Self_attention_block_27169963114663
// MI455X (gfx1250) — matrix operand order repaired and checked by reading, not compiled, not run on hardware
//
#include <hip/hip_runtime.h>

typedef _Float16 half_t;
typedef __attribute__((ext_vector_type(16))) _Float16 v16h;
typedef __attribute__((ext_vector_type(8)))  _Float16 v8h;
typedef __attribute__((ext_vector_type(4)))  _Float16 v4h;
typedef __attribute__((ext_vector_type(8)))  float    v8f;

#define BATCH   16
#define SEQ     2048
#define DMODEL  1024
#define SSTRIDE 2065   // 2048 + 17 pad (odd, 17%64 coprime-ish -> conflict-free row starts)

// A-fragment (16x32 f16, wave32): lane m=lane&15 is the row; hi=lane>>4 selects
// K-subrange {hi*8..hi*8+7} U {hi*8+16..hi*8+23} (ISA 7.12.2 16-bit A layout).
static __device__ __forceinline__ v16h make_afrag(const half_t* __restrict__ p) {
  v8h lo = *(const v8h*)p;
  v8h hi = *(const v8h*)(p + 16);
  return __builtin_shufflevector(lo, hi, 0,1,2,3,4,5,6,7,8,9,10,11,12,13,14,15);
}

// ---------------- conversion kernels ----------------
__global__ void cvt_f32_f16(const float* __restrict__ in, half_t* __restrict__ out, int n) {
  int i = (blockIdx.x * blockDim.x + threadIdx.x) * 4;
  if (i < n) {
    float4 v = *(const float4*)(in + i);
    v4h h; h[0] = (half_t)v.x; h[1] = (half_t)v.y; h[2] = (half_t)v.z; h[3] = (half_t)v.w;
    *(v4h*)(out + i) = h;
  }
}

// Wt[e][d] = (f16) W[d][e]
__global__ void transpose_cvt(const float* __restrict__ W, half_t* __restrict__ Wt) {
  int o = blockIdx.x * blockDim.x + threadIdx.x;   // o = e*D + d
  int e = o >> 10, d = o & (DMODEL - 1);
  Wt[o] = (half_t)W[d * DMODEL + e];
}

// ---------------- QKV projection: C[32768,1024] = Xf16 @ W (+bias) ----------------
// out_n: row-major f16 [B*S][D].  out_t (if non-null): transposed f16 Vt[b][d][s].
__global__ __launch_bounds__(256)
void qkv_gemm(const half_t* __restrict__ xb, const half_t* __restrict__ Wt,
              const float* __restrict__ bias,
              half_t* __restrict__ out_n, half_t* __restrict__ out_t) {
  int wave = threadIdx.x >> 5, lane = threadIdx.x & 31;
  int m = lane & 15, hi = lane >> 4;
  int mtile = blockIdx.x;
  int ntile = blockIdx.y * 8 + wave;

  const half_t* arow = xb + (size_t)(mtile * 16 + m) * DMODEL + hi * 8;
  const half_t* brow = Wt + (size_t)(ntile * 16 + m) * DMODEL + hi * 8;  // m doubles as n

  v8f c = {};
  for (int kt = 0; kt < DMODEL / 32; ++kt) {
    v16h a = make_afrag(arow + kt * 32);
    v16h b = make_afrag(brow + kt * 32);
    c = __builtin_amdgcn_wmma_f32_16x16x32_f16(false, a, false, b, (short)0, c, false, false);
  }

  int e = ntile * 16 + m;
  float bv = bias[e];
  int g0 = mtile * 16 + hi * 8;
#pragma unroll
  for (int r = 0; r < 8; ++r) {
    float v = c[r] + bv;
    int g = g0 + r;
    if (out_t) {
      int b_ = g >> 11, s_ = g & (SEQ - 1);
      out_t[(size_t)b_ * DMODEL * SEQ + (size_t)e * SEQ + s_] = (half_t)v;
    } else {
      out_n[(size_t)g * DMODEL + e] = (half_t)v;
    }
  }
}

// ---------------- attention: one block = (batch, 16 query rows) ----------------
#define ATTN_SMEM (16 * SSTRIDE * 4 + 16 * DMODEL * 2 + 16 * 4)

__global__ __launch_bounds__(256)
void attn_kernel(const half_t* __restrict__ Qb, const half_t* __restrict__ Kb,
                 const half_t* __restrict__ Vt, float* __restrict__ part) {
  int qt = blockIdx.x, bb = blockIdx.y;
  int tid = threadIdx.x, wave = tid >> 5, lane = tid & 31;
  int m = lane & 15, hi = lane >> 4;

  extern __shared__ char smem[];
  float*  Sb   = (float*)smem;                               // [16][SSTRIDE] f32 scores
  half_t* Qs   = (half_t*)(smem + 16 * SSTRIDE * 4);         // [16][DMODEL]
  float*  rsum = (float*)(smem + 16 * SSTRIDE * 4 + 16 * DMODEL * 2);

  // stage the 16xD query tile into LDS (f16, contiguous)
  const half_t* Qg = Qb + (size_t)(bb * SEQ + qt * 16) * DMODEL;
  for (int i = tid * 8; i < 16 * DMODEL; i += 256 * 8)
    *(v8h*)(Qs + i) = *(const v8h*)(Qg + i);
  __syncthreads();

  // ---- phase 1: S = (Q K^T) / sqrt(64); 128 key tiles split over 8 waves ----
  const half_t* Kg = Kb + (size_t)bb * SEQ * DMODEL;
  const half_t* aq = Qs + m * DMODEL + hi * 8;
  for (int kt = wave; kt < SEQ / 16; kt += 8) {
    const half_t* bk = Kg + (size_t)(kt * 16 + m) * DMODEL + hi * 8;  // B[n=s][k=d]
    v8f c = {};
    for (int dt = 0; dt < DMODEL / 32; ++dt) {
      v16h a = make_afrag(aq + dt * 32);
      v16h b = make_afrag(bk + dt * 32);
      c = __builtin_amdgcn_wmma_f32_16x16x32_f16(false, a, false, b, (short)0, c, false, false);
    }
#pragma unroll
    for (int r = 0; r < 8; ++r)
      Sb[(r + hi * 8) * SSTRIDE + kt * 16 + m] = c[r] * 0.125f;  // /sqrt(64)
  }
  __syncthreads();

  // ---- phase 2: row-wise softmax numerator (unnormalized), keep 1/rowsum ----
  {
    int row = tid >> 4, col0 = tid & 15;     // 16 threads per row, rows split by half-wave
    float* srow = Sb + row * SSTRIDE;
    float mx = -1e30f;
    for (int j = col0; j < SEQ; j += 16) mx = fmaxf(mx, srow[j]);
    for (int off = 8; off; off >>= 1) mx = fmaxf(mx, __shfl_xor(mx, off, 16));
    float sum = 0.f;
    for (int j = col0; j < SEQ; j += 16) {
      float e = __expf(srow[j] - mx);
      srow[j] = e;
      sum += e;
    }
    for (int off = 8; off; off >>= 1) sum += __shfl_xor(sum, off, 16);
    if (col0 == 0) rsum[row] = 1.0f / sum;
  }
  __syncthreads();

  // ---- phase 3: O = P @ V ; fold 1/rowsum into A-fragment conversion ----
  float rinv = rsum[m];   // each A-fragment lane holds exactly one P row
  const half_t* Vg = Vt + (size_t)bb * DMODEL * SEQ;
  const float* ap = Sb + m * SSTRIDE + hi * 8;
  for (int nt = wave; nt < DMODEL / 16; nt += 8) {
    const half_t* bv = Vg + (size_t)(nt * 16 + m) * SEQ + hi * 8;  // B[n=d][k=s], contiguous in s
    v8f c = {};
    for (int kt = 0; kt < SEQ / 32; ++kt) {
      const float* p0 = ap + kt * 32;
      v16h a;
#pragma unroll
      for (int j = 0; j < 8; ++j) {
        a[j]     = (half_t)(p0[j]      * rinv);
        a[8 + j] = (half_t)(p0[16 + j] * rinv);
      }
      v16h b = make_afrag(bv + kt * 32);
      c = __builtin_amdgcn_wmma_f32_16x16x32_f16(false, a, false, b, (short)0, c, false, false);
    }
    // deterministic pooled partial: sum over this tile's 16 query rows per column
    float v = c[0] + c[1] + c[2] + c[3] + c[4] + c[5] + c[6] + c[7];
    v += __shfl_xor(v, 16, 32);           // combine rows 0-7 with 8-15
    if (hi == 0)
      part[((size_t)(bb * (SEQ / 16) + qt)) * DMODEL + nt * 16 + m] = v;
  }
}

// ---------------- pooled[b][d] = mean over seq of weighted ----------------
__global__ void reduce_pool(const float* __restrict__ part, float* __restrict__ pooled) {
  int i = blockIdx.x * blockDim.x + threadIdx.x;  // i = b*D + d
  int b = i >> 10, d = i & (DMODEL - 1);
  const float* p = part + (size_t)b * (SEQ / 16) * DMODEL + d;
  float s = 0.f;
  for (int q = 0; q < SEQ / 16; ++q) s += p[(size_t)q * DMODEL];
  pooled[i] = s * (1.0f / (float)SEQ);
}

// ---------------- out[b][f] = pooled[b] @ Wo + bo (tiny f32 GEMV) ----------------
__global__ void out_proj(const float* __restrict__ pooled, const float* __restrict__ Wo,
                         const float* __restrict__ bo, float* __restrict__ out) {
  int f = blockIdx.x * blockDim.x + threadIdx.x;
  int b = blockIdx.y;
  const float* p = pooled + b * DMODEL;
  float acc = bo[f];
  for (int d = 0; d < DMODEL; ++d) acc += p[d] * Wo[(size_t)d * DMODEL + f];
  out[b * DMODEL + f] = acc;
}

extern "C" void kernel_launch(void* const* d_in, const int* in_sizes, int n_in,
                              void* d_out, int out_size, void* d_ws, size_t ws_size,
                              hipStream_t stream) {
  (void)in_sizes; (void)n_in; (void)out_size; (void)ws_size;
  const float* x  = (const float*)d_in[0];
  const float* Wq = (const float*)d_in[1];
  const float* bq = (const float*)d_in[2];
  const float* Wk = (const float*)d_in[3];
  const float* bk = (const float*)d_in[4];
  const float* Wv = (const float*)d_in[5];
  const float* bv = (const float*)d_in[6];
  const float* Wo = (const float*)d_in[7];
  const float* bo = (const float*)d_in[8];
  float* out = (float*)d_out;

  char* ws = (char*)d_ws;
  size_t off = 0;
  half_t* xb = (half_t*)(ws + off); off += (size_t)BATCH * SEQ * DMODEL * 2;        // 64 MB
  half_t* Wt = (half_t*)(ws + off); off += (size_t)3 * DMODEL * DMODEL * 2;         // 6 MB
  half_t* Qb = (half_t*)(ws + off); off += (size_t)BATCH * SEQ * DMODEL * 2;        // 64 MB
  half_t* Kb = (half_t*)(ws + off); off += (size_t)BATCH * SEQ * DMODEL * 2;        // 64 MB
  half_t* Vt = (half_t*)(ws + off); off += (size_t)BATCH * DMODEL * SEQ * 2;        // 64 MB
  float*  part   = (float*)(ws + off); off += (size_t)BATCH * (SEQ / 16) * DMODEL * 4; // 8 MB
  float*  pooled = (float*)(ws + off); off += (size_t)BATCH * DMODEL * 4;           // 64 KB

  const int nX = BATCH * SEQ * DMODEL;
  cvt_f32_f16<<<nX / (256 * 4), 256, 0, stream>>>(x, xb, nX);
  transpose_cvt<<<DMODEL * DMODEL / 256, 256, 0, stream>>>(Wq, Wt);
  transpose_cvt<<<DMODEL * DMODEL / 256, 256, 0, stream>>>(Wk, Wt + (size_t)DMODEL * DMODEL);
  transpose_cvt<<<DMODEL * DMODEL / 256, 256, 0, stream>>>(Wv, Wt + (size_t)2 * DMODEL * DMODEL);

  dim3 gq(BATCH * SEQ / 16, 8);
  qkv_gemm<<<gq, 256, 0, stream>>>(xb, Wt,                                bq, Qb, nullptr);
  qkv_gemm<<<gq, 256, 0, stream>>>(xb, Wt + (size_t)DMODEL * DMODEL,      bk, Kb, nullptr);
  qkv_gemm<<<gq, 256, 0, stream>>>(xb, Wt + (size_t)2 * DMODEL * DMODEL,  bv, nullptr, Vt);

  hipFuncSetAttribute((const void*)attn_kernel,
                      hipFuncAttributeMaxDynamicSharedMemorySize, (int)ATTN_SMEM);
  attn_kernel<<<dim3(SEQ / 16, BATCH), 256, ATTN_SMEM, stream>>>(Qb, Kb, Vt, part);

  reduce_pool<<<BATCH * DMODEL / 256, 256, 0, stream>>>(part, pooled);
  out_proj<<<dim3(DMODEL / 256, BATCH), 256, 0, stream>>>(pooled, Wo, bo, out);
}
